// HSTU_BSA_Triton_23484881174587
// MI455X (gfx1250) — matrix operand order repaired and checked by reading, not compiled, not run on hardware
//
#include <hip/hip_runtime.h>
#include <hip/hip_bf16.h>
#include <cstdint>

#define NSA_H  4
#define NSA_D  64
#define NSA_BS 32
#define NSA_S  4
#define NSA_NB 16   // max compressed blocks per sequence (seq_len <= 512)

typedef __attribute__((ext_vector_type(16))) _Float16 v16h;
typedef __attribute__((ext_vector_type(8)))  float    v8f;

// A-matrix (16x32 f16) per-lane element j -> K index (contiguous: K = 16*hal + j, matches B fills)
__device__ __forceinline__ int ka_idx(int j, int hal) {
  return j + hal * 16;
}
// SiLU via native transcendentals: v_exp_f32 + v_rcp_f32 (no IEEE div sequence)
__device__ __forceinline__ float silu_f(float x) {
  return x * __builtin_amdgcn_rcpf(1.0f + __expf(-x));
}

#define WMMA_F16(a, b, c) \
  __builtin_amdgcn_wmma_f32_16x16x32_f16(false, (a), false, (b), (short)0, (c), false, false)

// ---------------------------------------------------------------------------
// Kernel 1: block mean-pool of K and V -> k_cmp / v_cmp  (zero beyond seq end)
// grid: (NSA_NB, H, B), block: 64 threads (one per head-dim element)
// ---------------------------------------------------------------------------
__global__ void nsa_compress(const float* __restrict__ k, const float* __restrict__ v,
                             const long long* __restrict__ offs,
                             float* __restrict__ kc, float* __restrict__ vc) {
  const int blk = blockIdx.x, h = blockIdx.y, b = blockIdx.z;
  const int d = threadIdx.x;
  const long long off = offs[b];
  const int n = (int)(offs[b + 1] - off);
  float sk = 0.0f, sv = 0.0f;
  const int base = blk * NSA_BS;
  for (int i = 0; i < NSA_BS; ++i) {
    const int r = base + i;
    if (r < n) {
      const size_t idx = ((size_t)(off + r) * NSA_H + h) * NSA_D + d;
      sk += k[idx];
      sv += v[idx];
    }
  }
  const size_t o = (((size_t)b * NSA_H + h) * NSA_NB + blk) * NSA_D + d;
  kc[o] = sk * (1.0f / NSA_BS);
  vc[o] = sv * (1.0f / NSA_BS);
}

// ---------------------------------------------------------------------------
// Kernel 2: fused compressed attention + top-4 selection + selected attention.
// One wave32 per (q-tile of 16 rows, head, batch).
// ---------------------------------------------------------------------------
__global__ __launch_bounds__(32) void nsa_attn(
    const float* __restrict__ q, const float* __restrict__ k, const float* __restrict__ v,
    const float* __restrict__ gc, const float* __restrict__ gs,
    const long long* __restrict__ offs,
    const float* __restrict__ kc, const float* __restrict__ vc,
    float* __restrict__ ocmp, float* __restrict__ oslc) {

  const int qt = blockIdx.x, h = blockIdx.y, b = blockIdx.z;
  const long long off = offs[b];
  const int n = (int)(offs[b + 1] - off);
  if (qt * 16 >= n) return;                    // uniform: full wave exits together
  int cmpLen = (n + NSA_BS - 1) / NSA_BS;
  if (cmpLen > NSA_NB) cmpLen = NSA_NB;

  const int l   = threadIdx.x;
  const int hal = l >> 4;                      // lane half (K-split of fragments)
  const int ln  = l & 15;                      // M / N index within fragments
  const float scale = 0.125f;                  // 64^-0.5
  const float NEGBIG = -3.402823466e38f;

  __shared__ float    lds_s[16][16];           // causal-masked scores (for top-k)
  __shared__ float    lds_p[16][16];           // silu(scores) for compressed P
  __shared__ float    lds_ps[16][32];          // selected-branch P tile
  __shared__ unsigned lds_mask[16];            // per-row top-4 block bitmask
  __shared__ unsigned lds_uni;                 // union of row masks

  // ---- q A-fragments for this tile (M=16 rows, K=64 head dim in 2 chunks)
  const int qrow = qt * 16 + ln;
  const bool qvalid = qrow < n;
  const float* qp = q + ((size_t)(off + qrow) * NSA_H + h) * NSA_D;
  v16h aq0, aq1;
#pragma unroll
  for (int j = 0; j < 16; ++j) {
    const int dk = ka_idx(j, hal);
    aq0[j] = qvalid ? (_Float16)qp[dk]      : (_Float16)0.0f;
    aq1[j] = qvalid ? (_Float16)qp[dk + 32] : (_Float16)0.0f;
  }

  // ---- compressed scores: 16 rows x 16 blocks, K=64 -> 2 WMMAs
  const float* kcp = kc + (((size_t)b * NSA_H + h) * NSA_NB + ln) * NSA_D;
  v8f sc = {};
  {
    v16h bk0, bk1;
#pragma unroll
    for (int j = 0; j < 16; ++j) {
      const int dk = hal * 16 + j;             // B fragment: K = head dim
      bk0[j] = (_Float16)kcp[dk];
      bk1[j] = (_Float16)kcp[dk + 32];
    }
    sc = WMMA_F16(aq0, bk0, sc);
    sc = WMMA_F16(aq1, bk1, sc);
  }

#pragma unroll
  for (int r = 0; r < 8; ++r) {
    const int row = r + hal * 8;
    const int mg  = qt * 16 + row;
    const int qb  = mg >> 5;                   // query's block index
    const float s = sc[r] * scale;
    const bool caus = (ln <= qb);
    lds_s[row][ln] = caus ? s : NEGBIG;
    lds_p[row][ln] = caus ? silu_f(s) : 0.0f;
  }
  __syncthreads();

  // ---- per-row top-4 causal blocks (strict > == JAX lowest-index tie-break)
  if (l < 16) {
    unsigned m = 0;
#pragma unroll
    for (int t = 0; t < NSA_S; ++t) {
      float best = -3.0e38f;                   // excludes NEGBIG sentinel
      int bi = -1;
      for (int c2 = 0; c2 < NSA_NB; ++c2) {
        const float sv2 = lds_s[l][c2];
        if (!((m >> c2) & 1u) && sv2 > best) { best = sv2; bi = c2; }
      }
      if (bi >= 0) m |= 1u << bi;
    }
    lds_mask[l] = m;
  }
  __syncthreads();
  if (l == 0) {
    unsigned u = 0;
    for (int r2 = 0; r2 < 16; ++r2) u |= lds_mask[r2];
    lds_uni = u;
  }
  __syncthreads();

  // ---- o_cmp = P(16x16, K padded to 32) x v_cmp(16x64): 4 WMMAs
  v16h ap;
#pragma unroll
  for (int j = 0; j < 16; ++j) {
    const int kk = ka_idx(j, hal);
    ap[j] = (kk < 16) ? (_Float16)lds_p[ln][kk] : (_Float16)0.0f;
  }
  const float* vcp = vc + (((size_t)b * NSA_H + h) * NSA_NB) * NSA_D;
  v8f occ[4];
#pragma unroll
  for (int t = 0; t < 4; ++t) {
    v16h bv;
#pragma unroll
    for (int j = 0; j < 16; ++j) {
      const int kk = hal * 16 + j;             // K = compressed block index
      bv[j] = (kk < 16) ? (_Float16)vcp[(size_t)kk * NSA_D + t * 16 + ln]
                        : (_Float16)0.0f;
    }
    v8f z = {};
    occ[t] = WMMA_F16(ap, bv, z);
  }
#pragma unroll
  for (int r = 0; r < 8; ++r) {
    const int row = r + hal * 8;
    const int mg  = qt * 16 + row;
    if (mg < n) {
      const float g = gc[(size_t)(off + mg) * NSA_H + h];
      float* op = ocmp + ((size_t)(off + mg) * NSA_H + h) * NSA_D;
#pragma unroll
      for (int t = 0; t < 4; ++t) op[t * 16 + ln] = occ[t][r] * g;
    }
  }

  // ---- selected branch: iterate causal blocks, mask by per-row top-4 bits
  v8f oacc[4] = {};
  int qbMax = (qt * 16 + 15) >> 5;
  if (qbMax >= cmpLen) qbMax = cmpLen - 1;
  const unsigned uni = lds_uni;

  for (int kb = 0; kb <= qbMax; ++kb) {        // kb is wave-uniform
    if (!((uni >> kb) & 1u)) continue;
    const float* kbase = k + ((size_t)(off + kb * NSA_BS) * NSA_H + h) * NSA_D;
    const float* vbase = v + ((size_t)(off + kb * NSA_BS) * NSA_H + h) * NSA_D;
    __builtin_prefetch(vbase, 0, 1);           // global_prefetch for V block

    // scores vs 32 raw keys of this block (two 16-wide N tiles)
#pragma unroll
    for (int nt = 0; nt < 2; ++nt) {
      const int kpos = kb * NSA_BS + nt * 16 + ln;
      const bool kvld = kpos < n;
      const float* kp = kbase + (size_t)(nt * 16 + ln) * NSA_H * NSA_D;
      v16h bk0, bk1;
#pragma unroll
      for (int j = 0; j < 16; ++j) {
        const int dk = hal * 16 + j;
        bk0[j] = kvld ? (_Float16)kp[dk]      : (_Float16)0.0f;
        bk1[j] = kvld ? (_Float16)kp[dk + 32] : (_Float16)0.0f;
      }
      v8f sa = {};
      sa = WMMA_F16(aq0, bk0, sa);
      sa = WMMA_F16(aq1, bk1, sa);
#pragma unroll
      for (int r = 0; r < 8; ++r) {
        const int row = r + hal * 8;
        const int mg  = qt * 16 + row;
        const int kp2 = kb * NSA_BS + nt * 16 + ln;
        const bool ok = ((lds_mask[row] >> kb) & 1u) && (kp2 <= mg) && (kp2 < n);
        lds_ps[row][nt * 16 + ln] = ok ? silu_f(sa[r] * scale) : 0.0f;
      }
    }
    __syncthreads();

    // O += P(16x32) x V_block(32x64): 4 WMMAs (K=32 exactly)
    v16h aps;
#pragma unroll
    for (int j = 0; j < 16; ++j)
      aps[j] = (_Float16)lds_ps[ln][ka_idx(j, hal)];
#pragma unroll
    for (int t = 0; t < 4; ++t) {
      v16h bv;
#pragma unroll
      for (int j = 0; j < 16; ++j) {
        const int kk = hal * 16 + j;           // key index within block
        const int kp2 = kb * NSA_BS + kk;
        bv[j] = (kp2 < n) ? (_Float16)vbase[(size_t)kk * NSA_H * NSA_D + t * 16 + ln]
                          : (_Float16)0.0f;
      }
      oacc[t] = WMMA_F16(aps, bv, oacc[t]);
    }
    __syncthreads();                           // protect lds_ps WAR for next kb
  }

  // ---- store o_slc
#pragma unroll
  for (int r = 0; r < 8; ++r) {
    const int row = r + hal * 8;
    const int mg  = qt * 16 + row;
    if (mg < n) {
      const float g = gs[(size_t)(off + mg) * NSA_H + h];
      float* op = oslc + ((size_t)(off + mg) * NSA_H + h) * NSA_D;
#pragma unroll
      for (int t = 0; t < 4; ++t) op[t * 16 + ln] = oacc[t][r] * g;
    }
  }
}

// ---------------------------------------------------------------------------
extern "C" void kernel_launch(void* const* d_in, const int* in_sizes, int n_in,
                              void* d_out, int out_size, void* d_ws, size_t ws_size,
                              hipStream_t stream) {
  (void)n_in; (void)out_size; (void)ws_size;
  const float* q  = (const float*)d_in[0];
  const float* k  = (const float*)d_in[1];
  const float* v  = (const float*)d_in[2];
  const float* gc = (const float*)d_in[3];
  const float* gs = (const float*)d_in[4];
  const long long* offs = (const long long*)d_in[5];
  const int B = in_sizes[5] - 1;
  const int T = in_sizes[0] / (NSA_H * NSA_D);

  float* kc = (float*)d_ws;
  float* vc = kc + (size_t)B * NSA_H * NSA_NB * NSA_D;
  float* ocmp = (float*)d_out;
  float* oslc = ocmp + (size_t)T * NSA_H * NSA_D;

  nsa_compress<<<dim3(NSA_NB, NSA_H, B), NSA_D, 0, stream>>>(k, v, offs, kc, vc);
  nsa_attn<<<dim3((T + 15) / 16, NSA_H, B), 32, 0, stream>>>(
      q, k, v, gc, gs, offs, kc, vc, ocmp, oslc);
}
